// GNN_node_expander_79224966742698
// MI455X (gfx1250) — matrix operand order repaired and checked by reading, not compiled, not run on hardware
//
#include <hip/hip_runtime.h>

// ---- problem constants (from reference) ----
#define Nn   100000
#define Ee   500000
#define EEe  300000
#define Ll   3
#define Dd   128
#define EVv  5

typedef unsigned short ushort_t;
typedef __attribute__((ext_vector_type(16))) __bf16 v16bf;
typedef __attribute__((ext_vector_type(8)))  float  v8f;
typedef __attribute__((ext_vector_type(4)))  unsigned int u32x4;

union BfPack { u32x4 u[2]; v16bf v; };

__device__ __forceinline__ ushort_t f2bf(float f) {
  unsigned int u = __float_as_uint(f);
  u += 0x7FFFu + ((u >> 16) & 1u);          // round-to-nearest-even
  return (ushort_t)(u >> 16);
}

// ---------------- h init: h = (key_emb[x0] + val_emb[x1]) * mask ----------------
__global__ void k_hinit(const int* __restrict__ x, const int* __restrict__ mask,
                        const float* __restrict__ ke, const float* __restrict__ ve,
                        float* __restrict__ h) {
  int i = blockIdx.x * blockDim.x + threadIdx.x;
  int node = i >> 5;
  if (node >= Nn) return;
  int c = (i & 31) * 4;
  float mk = (float)mask[node];
  int a = x[node * 2 + 0], b = x[node * 2 + 1];
  const float* pa = ke + a * Dd + c;
  const float* pb = ve + b * Dd + c;
  float* ph = h + (size_t)node * Dd + c;
#pragma unroll
  for (int j = 0; j < 4; ++j) ph[j] = (pa[j] + pb[j]) * mk;
}

// ---------------- weight prep: f32 -> bf16, transposed to N-major ----------------
// W1[fl] is 128x256 (KxN) -> W1T[fl][n][k] ; W2[fl] is 256x128 -> W2T[fl][n][k]
__global__ void k_prepw(const float* __restrict__ W1, const float* __restrict__ W2,
                        ushort_t* __restrict__ W1T, ushort_t* __restrict__ W2T) {
  int i = blockIdx.x * blockDim.x + threadIdx.x;
  const int per = 128 * 256;
  if (i >= 9 * per) return;
  int fl = i / per, rem = i % per;
  int k1 = rem / 256, n1 = rem % 256;      // W1 element [fl][k1][n1]
  W1T[(size_t)fl * per + n1 * 128 + k1] = f2bf(W1[i]);
  int k2 = rem / 128, n2 = rem % 128;      // W2 element [fl][k2][n2]
  W2T[(size_t)fl * per + n2 * 256 + k2] = f2bf(W2[i]);
}

// ---------------- edge scatter: agg[dst] += relu(h[src] + eattr) ----------------
__global__ void k_edge(const float* __restrict__ h, float* __restrict__ agg,
                       const int* __restrict__ src, const int* __restrict__ dst, int ne,
                       const float* __restrict__ attr_tab, const int* __restrict__ eattr) {
  int i = blockIdx.x * blockDim.x + threadIdx.x;
  int e = i >> 5;
  if (e >= ne) return;
  int c = (i & 31) * 4;
  int s = src[e], d = dst[e];
  const float* ph = h + (size_t)s * Dd + c;
  float v[4];
#pragma unroll
  for (int j = 0; j < 4; ++j) v[j] = ph[j];
  if (attr_tab) {
    const float* pa = attr_tab + (size_t)eattr[e] * Dd + c;
#pragma unroll
    for (int j = 0; j < 4; ++j) v[j] += pa[j];
  }
  float* pg = agg + (size_t)d * Dd + c;
#pragma unroll
  for (int j = 0; j < 4; ++j) atomicAdd(pg + j, v[j] > 0.f ? v[j] : 0.f);
}

// ---------------- fused MLP: out = where(upd, relu(z@W1+b1)@W2+b2, h) ----------------
// z = (1+eps)*h + agg ; also accumulates BN sum / sumsq per column into stats[0..255]
#define AP1 136   // LDS row stride (bf16 elems) for 64x128 A1  (pad vs bank conflicts)
#define AP2 264   // LDS row stride (bf16 elems) for 64x256 A2
__global__ __launch_bounds__(256)
void k_mlp(const float* __restrict__ h, const float* __restrict__ agg,
           const float* __restrict__ b1, const float* __restrict__ b2,
           const ushort_t* __restrict__ W1T, const ushort_t* __restrict__ W2T,
           const int* __restrict__ nodemask, int updmode,
           const float* __restrict__ eps_ptr, int eps_idx,
           float* __restrict__ outbuf, float* __restrict__ stats) {
  __shared__ ushort_t A1[64 * AP1];
  __shared__ ushort_t A2[64 * AP2];
  const int tid = threadIdx.x;
  const int base = blockIdx.x * 64;
  const float epsv = 1.0f + eps_ptr[eps_idx];

  // phase 0: build bf16 A1 = (1+eps)*h + agg  (zero-pad rows >= N)
  for (int i = tid; i < 64 * 128; i += 256) {
    int r = i >> 7, c = i & 127;
    int row = base + r;
    float z = 0.f;
    if (row < Nn) {
      size_t o = (size_t)row * Dd + c;
      z = epsv * h[o] + agg[o];
    }
    A1[r * AP1 + c] = f2bf(z);
  }
  __syncthreads();

  const int wave = tid >> 5, lane = tid & 31;
  const int lh = lane >> 4, lx = lane & 15;

  // GEMM1: (64x128) x (128x256) -> relu(+b1) -> A2 bf16.  Wave owns ntiles {2w,2w+1}.
  for (int nti = 0; nti < 2; ++nti) {
    int nt = 2 * wave + nti;
    int col = nt * 16 + lx;
    float bias = b1[col];
#pragma unroll
    for (int mt = 0; mt < 4; ++mt) {
      v8f acc = {};
#pragma unroll
      for (int kt = 0; kt < 4; ++kt) {
        BfPack a, b;
        int ar = (mt * 16 + lx) * AP1 + kt * 32 + 8 * lh;
        a.u[0] = *(const u32x4*)&A1[ar];
        a.u[1] = *(const u32x4*)&A1[ar + 16];
        const ushort_t* wp = W1T + (size_t)col * 128 + kt * 32 + 8 * lh;
        b.u[0] = ((const u32x4*)wp)[0];
        b.u[1] = ((const u32x4*)wp)[2];
        acc = __builtin_amdgcn_wmma_f32_16x16x32_bf16(false, a.v, false, b.v,
                                                      (short)0, acc, false, false);
      }
#pragma unroll
      for (int r = 0; r < 8; ++r) {
        int row = mt * 16 + r + 8 * lh;
        float v = acc[r] + bias;
        A2[row * AP2 + col] = f2bf(v > 0.f ? v : 0.f);
      }
    }
  }
  __syncthreads();

  // GEMM2: (64x256) x (256x128) + b2 -> where(upd) -> outbuf + BN partials.
  {
    int nt = wave;                 // 8 waves cover 128 output cols
    int col = nt * 16 + lx;
    float bias = b2[col];
    float s1 = 0.f, s2 = 0.f;
#pragma unroll
    for (int mt = 0; mt < 4; ++mt) {
      v8f acc = {};
#pragma unroll
      for (int kt = 0; kt < 8; ++kt) {
        BfPack a, b;
        int ar = (mt * 16 + lx) * AP2 + kt * 32 + 8 * lh;
        a.u[0] = *(const u32x4*)&A2[ar];
        a.u[1] = *(const u32x4*)&A2[ar + 16];
        const ushort_t* wp = W2T + (size_t)col * 256 + kt * 32 + 8 * lh;
        b.u[0] = ((const u32x4*)wp)[0];
        b.u[1] = ((const u32x4*)wp)[2];
        acc = __builtin_amdgcn_wmma_f32_16x16x32_bf16(false, a.v, false, b.v,
                                                      (short)0, acc, false, false);
      }
#pragma unroll
      for (int r = 0; r < 8; ++r) {
        int row = base + mt * 16 + r + 8 * lh;
        if (row < Nn) {
          float v = acc[r] + bias;
          int mk = nodemask[row];
          bool upd = updmode ? (mk == 0) : (mk > 0);
          float hv = h[(size_t)row * Dd + col];
          float o = upd ? v : hv;
          outbuf[(size_t)row * Dd + col] = o;
          s1 += o; s2 += o * o;
        }
      }
    }
    s1 += __shfl_xor(s1, 16, 32);
    s2 += __shfl_xor(s2, 16, 32);
    if (lh == 0) {
      atomicAdd(&stats[col], s1);
      atomicAdd(&stats[Dd + col], s2);
    }
  }
}

// ---------------- BN finalize: per-column scale/shift ----------------
__global__ void k_bnfin(const float* __restrict__ stats, const float* __restrict__ g,
                        const float* __restrict__ b, float* __restrict__ ss) {
  int c = threadIdx.x;
  const float inv_n = 1.0f / (float)Nn;
  float m = stats[c] * inv_n;
  float var = stats[Dd + c] * inv_n - m * m;
  float sc = g[c] * rsqrtf(var + 1e-5f);
  ss[c] = sc;
  ss[Dd + c] = b[c] - m * sc;
}

// ---------------- BN apply + residual: h += relu(z*scale + shift) ----------------
__global__ void k_bnapply(float* __restrict__ h, const float* __restrict__ ob,
                          const float* __restrict__ ss) {
  int i = blockIdx.x * blockDim.x + threadIdx.x;
  size_t base = (size_t)i * 4;
  if (base >= (size_t)Nn * Dd) return;
  int c0 = (int)(base & (Dd - 1));
#pragma unroll
  for (int j = 0; j < 4; ++j) {
    float z = ob[base + j];
    float v = z * ss[c0 + j] + ss[Dd + c0 + j];
    h[base + j] += (v > 0.f ? v : 0.f);
  }
}

// ---------------- LayerNorm over D (wave per row) ----------------
__global__ void k_ln(const float* __restrict__ hin, float* __restrict__ hout,
                     const float* __restrict__ g, const float* __restrict__ b) {
  int wave = threadIdx.x >> 5, lane = threadIdx.x & 31;
  int row = blockIdx.x * 8 + wave;
  if (row >= Nn) return;
  const float* p = hin + (size_t)row * Dd + lane * 4;
  float v[4], s1 = 0.f, s2 = 0.f;
#pragma unroll
  for (int j = 0; j < 4; ++j) { v[j] = p[j]; s1 += v[j]; s2 += v[j] * v[j]; }
  for (int o = 16; o > 0; o >>= 1) { s1 += __shfl_xor(s1, o, 32); s2 += __shfl_xor(s2, o, 32); }
  float m = s1 * (1.0f / Dd);
  float var = s2 * (1.0f / Dd) - m * m;
  float rs = rsqrtf(var + 1e-5f);
  float* q = hout + (size_t)row * Dd + lane * 4;
#pragma unroll
  for (int j = 0; j < 4; ++j) {
    int c = lane * 4 + j;
    q[j] = (v[j] - m) * rs * g[c] + b[c];
  }
}

extern "C" void kernel_launch(void* const* d_in, const int* in_sizes, int n_in,
                              void* d_out, int out_size, void* d_ws, size_t ws_size,
                              hipStream_t stream) {
  (void)in_sizes; (void)n_in; (void)out_size; (void)ws_size;
  const int*   x    = (const int*)d_in[0];
  const int*   ei   = (const int*)d_in[1];
  const int*   ea   = (const int*)d_in[2];
  const int*   xei  = (const int*)d_in[3];
  const int*   msk  = (const int*)d_in[4];
  const float* ke   = (const float*)d_in[5];
  const float* ve   = (const float*)d_in[6];
  const float* eemb = (const float*)d_in[7];
  const float* eps  = (const float*)d_in[8];
  const float* W1   = (const float*)d_in[9];
  const float* b1   = (const float*)d_in[10];
  const float* W2   = (const float*)d_in[11];
  const float* b2   = (const float*)d_in[12];
  const float* bng  = (const float*)d_in[13];
  const float* bnb  = (const float*)d_in[14];
  const float* lng  = (const float*)d_in[15];
  const float* lnb  = (const float*)d_in[16];
  float* out = (float*)d_out;

  // workspace carve
  char* ws = (char*)d_ws;
  const size_t HB = (size_t)Nn * Dd * sizeof(float);      // 51.2 MB
  float* h     = (float*)(ws);
  float* agg   = (float*)(ws + HB);
  float* ob    = (float*)(ws + 2 * HB);
  float* stats = (float*)(ws + 3 * HB);                    // 256 floats
  float* ss    = stats + 256;                              // 256 floats
  ushort_t* W1T = (ushort_t*)(ws + 3 * HB + 4096);
  ushort_t* W2T = W1T + (size_t)9 * 128 * 256;

  k_hinit<<<(Nn * 32 + 255) / 256, 256, 0, stream>>>(x, msk, ke, ve, h);
  k_prepw<<<(9 * 128 * 256 + 255) / 256, 256, 0, stream>>>(W1, W2, W1T, W2T);

  for (int l = 0; l < Ll; ++l) {
    for (int f = 0; f < 3; ++f) {
      const int* src; const int* dst;
      const float* at = nullptr; const int* eat = nullptr; int ne;
      if (f == 0)      { src = ei;        dst = ei + Ee;   at = eemb + (size_t)l * EVv * Dd; eat = ea; ne = Ee; }
      else if (f == 1) { src = xei;       dst = xei + EEe; ne = EEe; }
      else             { src = xei + EEe; dst = xei;       ne = EEe; }   // reversed expander

      hipMemsetAsync(agg, 0, HB, stream);
      hipMemsetAsync(stats, 0, 256 * sizeof(float), stream);

      k_edge<<<(ne * 32 + 255) / 256, 256, 0, stream>>>(h, agg, src, dst, ne, at, eat);

      int fl = f * Ll + l;
      int updmode = (f == 1) ? 1 : 0;
      k_mlp<<<(Nn + 63) / 64, 256, 0, stream>>>(h, agg,
            b1 + (size_t)fl * 2 * Dd, b2 + (size_t)fl * Dd,
            W1T + (size_t)fl * 128 * 256, W2T + (size_t)fl * 128 * 256,
            msk, updmode, eps, fl, ob, stats);

      k_bnfin<<<1, Dd, 0, stream>>>(stats, bng + (size_t)fl * Dd, bnb + (size_t)fl * Dd, ss);
      k_bnapply<<<(Nn * Dd / 4 + 255) / 256, 256, 0, stream>>>(h, ob, ss);
    }
    k_ln<<<(Nn + 7) / 8, 256, 0, stream>>>(h, (l == Ll - 1) ? out : h,
                                           lng + (size_t)l * Dd, lnb + (size_t)l * Dd);
  }
}
